// MTCNN_79972291052305
// MI455X (gfx1250) — matrix operand order repaired and checked by reading, not compiled, not run on hardware
//
#include <hip/hip_runtime.h>
#include <cstdint>

typedef __attribute__((ext_vector_type(16))) _Float16 v16h;
typedef __attribute__((ext_vector_type(8)))  float    v8f;

// ---------------------------------------------------------------------------
// B fragment loader: weights pre-packed so lane L reads dwords [base + j*32 + L]
// (coalesced).  Packing already encodes the ISA B layout (col = lane&15,
// K half = lane>>4, VGPR j holds K pair {k0, k0+1}).
// ---------------------------------------------------------------------------
__device__ __forceinline__ v16h load_b_frag(const uint32_t* __restrict__ bp, int base) {
  const int lane = threadIdx.x & 31;
  union { uint32_t u[8]; v16h h; } r;
#pragma unroll
  for (int j = 0; j < 8; ++j) r.u[j] = bp[base + j * 32 + lane];
  return r.h;
}

// A-fragment K index inside a 32-wide k-step for vgpr j, given lane half.
__device__ __forceinline__ int a_kk(int j, int half) {
  return ((j >> 2) << 4) + (half << 3) + ((j & 3) << 1);
}

// Pack two f32 into an f16-pair dword for LDS stores.
__device__ __forceinline__ uint32_t pk_f16(float v0, float v1) {
  union { _Float16 h[2]; uint32_t u; } p;
  p.h[0] = (_Float16)v0; p.h[1] = (_Float16)v1;
  return p.u;
}

// Async global -> LDS 16-byte copy (CDNA5 TDM-lite path, ASYNCcnt tracked).
// LDS byte address = low 32 bits of the generic pointer (flat aperture rule).
__device__ __forceinline__ void async_copy_b128(const void* gptr, void* lptr) {
  const uint32_t lds = (uint32_t)(uintptr_t)lptr;
  const uint64_t ga  = (uint64_t)(uintptr_t)gptr;
  asm volatile("global_load_async_to_lds_b128 %0, %1, off" :: "v"(lds), "v"(ga) : "memory");
}
__device__ __forceinline__ void async_wait0() {
  asm volatile("s_wait_asynccnt 0" ::: "memory");
}

// ---------------------------------------------------------------------------
// Pack W[OC][K] (f32, row-major) into WMMA-B fragment order (f16 pairs).
// grid = ntiles*ksteps, block = 256.
// ---------------------------------------------------------------------------
__global__ void __launch_bounds__(256)
pack_b_kernel(const float* __restrict__ w, uint32_t* __restrict__ dst,
              int OC, int K, int ksteps) {
  const int nt = blockIdx.x / ksteps, s = blockIdx.x % ksteps;
  const int j = threadIdx.x >> 5, lane = threadIdx.x & 31;
  const int n = nt * 16 + (lane & 15);
  const int half = lane >> 4;
  const int k0 = s * 32 + ((j >> 2) << 4) + half * 8 + ((j & 3) << 1);
  _Float16 e0 = (_Float16)0.0f, e1 = (_Float16)0.0f;
  if (n < OC) {
    if (k0 < K)     e0 = (_Float16)w[(size_t)n * K + k0];
    if (k0 + 1 < K) e1 = (_Float16)w[(size_t)n * K + k0 + 1];
  }
  union { _Float16 h[2]; uint32_t u; } p;
  p.h[0] = e0; p.h[1] = e1;
  dst[(size_t)blockIdx.x * 256 + j * 32 + lane] = p.u;
}

// ---------------------------------------------------------------------------
// conv1 3->32 (3x3) + PReLU + maxpool 3x3 s2 ceil (46 -> 23), banded.
// grid = N*3 (3 pool-row bands of 8/8/7), block = 256.
// ---------------------------------------------------------------------------
__global__ void __launch_bounds__(256)
conv1_pool_kernel(const float* __restrict__ x, const uint32_t* __restrict__ wp,
                  const float* __restrict__ bias, const float* __restrict__ alpha,
                  _Float16* __restrict__ out) {
  __shared__ _Float16 sin[3600];          // 3*19*48 = 2736 valid + zero pad (K taps 27..31)
  __shared__ _Float16 sout[32 * 784];     // conv band, stride padded to 784 (49 tiles)
  const int n    = blockIdx.x / 3;
  const int band = blockIdx.x % 3;
  const int tid  = threadIdx.x;
  const int pr0  = band * 8;
  const int nrp  = (band < 2) ? 8 : 7;
  const int cr0  = pr0 * 2;
  const int ncr  = (46 - cr0 < 17) ? (46 - cr0) : 17;           // conv rows in band
  const int nir  = (48 - cr0 < ncr + 2) ? (48 - cr0) : ncr + 2; // input rows
  const int M    = ncr * 46;

  const float* xin = x + (size_t)n * 6912;
  for (int i = tid; i < 3600; i += 256) {
    _Float16 v = (_Float16)0.0f;
    if (i < 2736) {
      const int ic = i / 912, rem = i - ic * 912;
      const int r = rem / 48, cidx = rem - r * 48;
      if (r < nir) v = (_Float16)xin[ic * 2304 + (cr0 + r) * 48 + cidx];
    }
    sin[i] = v;
  }
  __syncthreads();

  const int wave = tid >> 5, lane = tid & 31;
  const int half = lane >> 4, l15 = lane & 15;
  const int mtiles = (M + 15) >> 4;
  for (int t = wave; t < mtiles * 2; t += 8) {
    const int mt = t >> 1, nt = t & 1;
    const int m  = mt * 16 + l15;
    const int mc = m < M ? m : M - 1;        // clamp: padded C rows are junk, never read
    const int oh = mc / 46, ow = mc - oh * 46;
    v16h a;
#pragma unroll
    for (int j = 0; j < 8; ++j) {
      const int kk = a_kk(j, half);          // K = 27 padded to 32; taps >=27 read zero pad
      int ic = kk / 9, r9 = kk - ic * 9;
      const _Float16 e0 = sin[ic * 912 + (oh + r9 / 3) * 48 + ow + (r9 % 3)];
      ic = (kk + 1) / 9; r9 = (kk + 1) - ic * 9;
      const _Float16 e1 = sin[ic * 912 + (oh + r9 / 3) * 48 + ow + (r9 % 3)];
      a[2 * j] = e0; a[2 * j + 1] = e1;
    }
    v16h b = load_b_frag(wp, nt * 256);
    v8f c = {};
    c = __builtin_amdgcn_wmma_f32_16x16x32_f16(false, a, false, b, (short)0, c, false, false);
    const int ch = nt * 16 + l15;
    const float bi = bias[ch], al = alpha[ch];
    uint32_t* sp = (uint32_t*)&sout[ch * 784 + mt * 16 + half * 8];
#pragma unroll
    for (int i = 0; i < 4; ++i) {
      float v0 = c[2 * i] + bi;     v0 = v0 >= 0.0f ? v0 : al * v0;
      float v1 = c[2 * i + 1] + bi; v1 = v1 >= 0.0f ? v1 : al * v1;
      sp[i] = pk_f16(v0, v1);
    }
  }
  __syncthreads();

  _Float16* op = out + (size_t)n * 16928;   // [32][23][23]
  const int nout = 32 * nrp * 23;
  for (int i = tid; i < nout; i += 256) {
    const int ch = i / (nrp * 23), rem = i - ch * (nrp * 23);
    const int lph = rem / 23, pw = rem - lph * 23;
    float mx = -3.0e38f;
    for (int dh = 0; dh < 3; ++dh) {
      const int lr = lph * 2 + dh;
      if (lr >= ncr) break;                 // bottom ceil-pad (-inf)
      for (int dw = 0; dw < 3; ++dw) {
        const int ww = pw * 2 + dw;
        if (ww >= 46) break;                // right ceil-pad (-inf)
        const float v = (float)sout[ch * 784 + lr * 46 + ww];
        mx = v > mx ? v : mx;
      }
    }
    op[ch * 529 + (pr0 + lph) * 23 + pw] = (_Float16)mx;
  }
}

// ---------------------------------------------------------------------------
// conv2 32->64 (3x3, 23->21) + PReLU + pool 3x3 s2 (21->10), 2 bands of 5 rows.
// grid = N*2, block 256.
// ---------------------------------------------------------------------------
__global__ void __launch_bounds__(256)
conv2_pool_kernel(const _Float16* __restrict__ in, const uint32_t* __restrict__ wp,
                  const float* __restrict__ bias, const float* __restrict__ alpha,
                  _Float16* __restrict__ out) {
  __shared__ _Float16 sin[32 * 13 * 23];  // input band, stride 299/ch
  __shared__ _Float16 sout[64 * 240];     // conv band, stride padded to 240 (15 tiles)
  const int n = blockIdx.x >> 1, band = blockIdx.x & 1;
  const int tid = threadIdx.x;
  const int pb0 = band * 5;               // pool rows [pb0, pb0+5)
  const int cr0 = pb0 * 2;                // conv rows [cr0, cr0+11)

  const _Float16* ip = in + (size_t)n * 16928;
  for (int i = tid; i < 32 * 13 * 23; i += 256) {
    const int ic = i / 299, rem = i - ic * 299;
    const int r = rem / 23, cidx = rem - r * 23;
    sin[i] = ip[ic * 529 + (cr0 + r) * 23 + cidx];
  }
  __syncthreads();

  const int wave = tid >> 5, lane = tid & 31;
  const int half = lane >> 4, l15 = lane & 15;
  const int M = 231;
  for (int t = wave; t < 15 * 4; t += 8) {
    const int mt = t >> 2, nt = t & 3;
    const int m  = mt * 16 + l15;
    const int mc = m < M ? m : M - 1;
    const int oh = mc / 21, ow = mc - oh * 21;
    v8f c = {};
    for (int s = 0; s < 9; ++s) {         // K = 288 = 9 * 32 (exact)
      v16h a;
#pragma unroll
      for (int j = 0; j < 8; ++j) {
        const int kk = s * 32 + a_kk(j, half);
        int ic = kk / 9, r9 = kk - ic * 9;
        const _Float16 e0 = sin[ic * 299 + (oh + r9 / 3) * 23 + ow + (r9 % 3)];
        ic = (kk + 1) / 9; r9 = (kk + 1) - ic * 9;
        const _Float16 e1 = sin[ic * 299 + (oh + r9 / 3) * 23 + ow + (r9 % 3)];
        a[2 * j] = e0; a[2 * j + 1] = e1;
      }
      v16h b = load_b_frag(wp, (nt * 9 + s) * 256);
      c = __builtin_amdgcn_wmma_f32_16x16x32_f16(false, a, false, b, (short)0, c, false, false);
    }
    const int ch = nt * 16 + l15;
    const float bi = bias[ch], al = alpha[ch];
    uint32_t* sp = (uint32_t*)&sout[ch * 240 + mt * 16 + half * 8];
#pragma unroll
    for (int i = 0; i < 4; ++i) {
      float v0 = c[2 * i] + bi;     v0 = v0 >= 0.0f ? v0 : al * v0;
      float v1 = c[2 * i + 1] + bi; v1 = v1 >= 0.0f ? v1 : al * v1;
      sp[i] = pk_f16(v0, v1);
    }
  }
  __syncthreads();

  _Float16* op = out + (size_t)n * 6400;  // [64][10][10]
  for (int i = tid; i < 64 * 50; i += 256) {
    const int ch = i / 50, rem = i - ch * 50;
    const int lp = rem / 10, pw = rem - lp * 10;
    float mx = -3.0e38f;
#pragma unroll
    for (int dh = 0; dh < 3; ++dh)
#pragma unroll
      for (int dw = 0; dw < 3; ++dw) {
        const float v = (float)sout[ch * 240 + (lp * 2 + dh) * 21 + pw * 2 + dw];
        mx = v > mx ? v : mx;
      }
    op[ch * 100 + (pb0 + lp) * 10 + pw] = (_Float16)mx;
  }
}

// ---------------------------------------------------------------------------
// conv3 64->64 (3x3, 10->8) + PReLU + pool 2x2 (8->4) + conv4 64->128 (2x2,
// 4->3) + PReLU + permuted flatten [n][w*384 + h*128 + c].  grid = N.
// ---------------------------------------------------------------------------
__global__ void __launch_bounds__(256)
conv34_kernel(const _Float16* __restrict__ in,
              const uint32_t* __restrict__ w3p, const float* __restrict__ b3,
              const float* __restrict__ a3,
              const uint32_t* __restrict__ w4p, const float* __restrict__ b4,
              const float* __restrict__ a4,
              _Float16* __restrict__ flat) {
  __shared__ __attribute__((aligned(16))) _Float16 sin[64 * 100];
  __shared__ _Float16 sconv[64 * 64];
  __shared__ _Float16 spool[64 * 16];
  const int n = blockIdx.x, tid = threadIdx.x;
  const _Float16* ip = in + (size_t)n * 6400;
  // async DMA stage: 6400 f16 = 800 x b128
  for (int i = tid; i < 800; i += 256)
    async_copy_b128(ip + (size_t)i * 8, &sin[i * 8]);
  async_wait0();
  __syncthreads();

  const int wave = tid >> 5, lane = tid & 31;
  const int half = lane >> 4, l15 = lane & 15;

  // conv3: M=64 (8x8), N=64, K=576 = 18*32
  for (int t = wave; t < 16; t += 8) {
    const int mt = t >> 2, nt = t & 3;
    const int m = mt * 16 + l15;
    const int oh = m >> 3, ow = m & 7;
    v8f c = {};
    for (int s = 0; s < 18; ++s) {
      v16h a;
#pragma unroll
      for (int j = 0; j < 8; ++j) {
        const int kk = s * 32 + a_kk(j, half);
        int ic = kk / 9, r9 = kk - ic * 9;
        const _Float16 e0 = sin[ic * 100 + (oh + r9 / 3) * 10 + ow + (r9 % 3)];
        ic = (kk + 1) / 9; r9 = (kk + 1) - ic * 9;
        const _Float16 e1 = sin[ic * 100 + (oh + r9 / 3) * 10 + ow + (r9 % 3)];
        a[2 * j] = e0; a[2 * j + 1] = e1;
      }
      v16h b = load_b_frag(w3p, (nt * 18 + s) * 256);
      c = __builtin_amdgcn_wmma_f32_16x16x32_f16(false, a, false, b, (short)0, c, false, false);
    }
    const int ch = nt * 16 + l15;
    const float bi = b3[ch], al = a3[ch];
    uint32_t* sp = (uint32_t*)&sconv[ch * 64 + mt * 16 + half * 8];
#pragma unroll
    for (int i = 0; i < 4; ++i) {
      float v0 = c[2 * i] + bi;     v0 = v0 >= 0.0f ? v0 : al * v0;
      float v1 = c[2 * i + 1] + bi; v1 = v1 >= 0.0f ? v1 : al * v1;
      sp[i] = pk_f16(v0, v1);
    }
  }
  __syncthreads();

  // pool 2x2 s2: 8 -> 4
  for (int i = tid; i < 64 * 16; i += 256) {
    const int ch = i >> 4, rem = i & 15;
    const int ph = rem >> 2, pw = rem & 3;
    const float v00 = (float)sconv[ch * 64 + (ph * 2) * 8 + pw * 2];
    const float v01 = (float)sconv[ch * 64 + (ph * 2) * 8 + pw * 2 + 1];
    const float v10 = (float)sconv[ch * 64 + (ph * 2 + 1) * 8 + pw * 2];
    const float v11 = (float)sconv[ch * 64 + (ph * 2 + 1) * 8 + pw * 2 + 1];
    float mx = v00 > v01 ? v00 : v01;
    mx = v10 > mx ? v10 : mx;
    mx = v11 > mx ? v11 : mx;
    spool[i] = (_Float16)mx;
  }
  __syncthreads();

  // conv4: M=9 (pad 16), N=128 (wave = ntile), K=256 = 8*32
  {
    const int nt = wave;
    const int mc = l15 < 9 ? l15 : 8;       // clamp A rows; store is guarded below
    const int oh = mc / 3, ow = mc - oh * 3;
    v8f c = {};
    for (int s = 0; s < 8; ++s) {
      v16h a;
#pragma unroll
      for (int j = 0; j < 8; ++j) {
        const int kk = s * 32 + a_kk(j, half);
        int ic = kk >> 2, r4 = kk & 3;
        const _Float16 e0 = spool[ic * 16 + (oh + (r4 >> 1)) * 4 + ow + (r4 & 1)];
        ic = (kk + 1) >> 2; r4 = (kk + 1) & 3;
        const _Float16 e1 = spool[ic * 16 + (oh + (r4 >> 1)) * 4 + ow + (r4 & 1)];
        a[2 * j] = e0; a[2 * j + 1] = e1;
      }
      v16h b = load_b_frag(w4p, (nt * 8 + s) * 256);
      c = __builtin_amdgcn_wmma_f32_16x16x32_f16(false, a, false, b, (short)0, c, false, false);
    }
    const int ch = nt * 16 + l15;
    const float bi = b4[ch], al = a4[ch];
    _Float16* fp = flat + (size_t)n * 1152;
#pragma unroll
    for (int i = 0; i < 8; ++i) {
      const int mm = half * 8 + i;
      if (mm < 9) {
        float v = c[i] + bi;
        v = v >= 0.0f ? v : al * v;
        const int hh = mm / 3, wi = mm - hh * 3;
        fp[wi * 384 + hh * 128 + ch] = (_Float16)v;  // permute (0,3,2,1) flatten
      }
    }
  }
}

// ---------------------------------------------------------------------------
// dense5: [N,1152] x [1152,256] + bias + PReLU -> feat f16.  One wave per
// 16x16 tile, K = 36 steps.  grid = N/8, block 256.
// ---------------------------------------------------------------------------
__global__ void __launch_bounds__(256)
dense5_kernel(const _Float16* __restrict__ flat, const uint32_t* __restrict__ wp,
              const float* __restrict__ bias, const float* __restrict__ alpha,
              _Float16* __restrict__ feat) {
  const int tid = threadIdx.x;
  const int wave = tid >> 5, lane = tid & 31;
  const int half = lane >> 4, l15 = lane & 15;
  const int t = blockIdx.x * 8 + wave;
  const int mt = t >> 4, nt = t & 15;
  const uint32_t* a32 = (const uint32_t*)flat;
  const int m = mt * 16 + l15;
  v8f c = {};
  for (int s = 0; s < 36; ++s) {
    union { uint32_t u[8]; v16h h; } ua;
#pragma unroll
    for (int j = 0; j < 8; ++j) {
      const int kk = s * 32 + a_kk(j, half);     // even -> one dword = f16 pair
      ua.u[j] = a32[((size_t)m * 1152 + kk) >> 1];
    }
    v16h b = load_b_frag(wp, (nt * 36 + s) * 256);
    c = __builtin_amdgcn_wmma_f32_16x16x32_f16(false, ua.h, false, b, (short)0, c, false, false);
  }
  const int ch = nt * 16 + l15;
  const float bi = bias[ch], al = alpha[ch];
#pragma unroll
  for (int i = 0; i < 8; ++i) {
    float v = c[i] + bi;
    v = v >= 0.0f ? v : al * v;
    feat[(size_t)(mt * 16 + half * 8 + i) * 256 + ch] = (_Float16)v;
  }
}

// ---------------------------------------------------------------------------
// heads: 256 -> (2 softmax, 4, 10).  16 rows/block, thread = (row, slot).
// d_out layout: b [N*4] | c [N*10] | a [N*2]  (reference returns (b, c, a)).
// ---------------------------------------------------------------------------
__global__ void __launch_bounds__(256)
heads_kernel(const _Float16* __restrict__ feat,
             const float* __restrict__ w61, const float* __restrict__ b61,
             const float* __restrict__ w62, const float* __restrict__ b62,
             const float* __restrict__ w63, const float* __restrict__ b63,
             float* __restrict__ outp, int ofsC, int ofsA) {
  __shared__ __attribute__((aligned(16))) _Float16 sf[16 * 256];
  const int tid = threadIdx.x;
  const int r0 = blockIdx.x * 16;
  // async DMA stage: 4096 f16 = 512 x b128
  for (int i = tid; i < 512; i += 256)
    async_copy_b128(feat + (size_t)r0 * 256 + (size_t)i * 8, &sf[i * 8]);
  async_wait0();
  __syncthreads();
  const int lr = tid >> 4, slot = tid & 15;
  const int row = r0 + lr;
  const _Float16* f = sf + lr * 256;
  if (slot < 4) {
    const float* w = w62 + slot * 256;
    float acc = b62[slot];
    for (int k = 0; k < 256; ++k) acc += (float)f[k] * w[k];
    outp[row * 4 + slot] = acc;
  } else if (slot < 14) {
    const int o = slot - 4;
    const float* w = w63 + o * 256;
    float acc = b63[o];
    for (int k = 0; k < 256; ++k) acc += (float)f[k] * w[k];
    outp[ofsC + row * 10 + o] = acc;
  } else if (slot == 14) {
    float l0 = b61[0], l1 = b61[1];
    for (int k = 0; k < 256; ++k) {
      const float fv = (float)f[k];
      l0 += fv * w61[k];
      l1 += fv * w61[256 + k];
    }
    const float mxl = l0 > l1 ? l0 : l1;
    const float e0 = __expf(l0 - mxl), e1 = __expf(l1 - mxl);
    const float inv = 1.0f / (e0 + e1);
    outp[ofsA + row * 2 + 0] = e0 * inv;
    outp[ofsA + row * 2 + 1] = e1 * inv;
  }
}

// ---------------------------------------------------------------------------
extern "C" void kernel_launch(void* const* d_in, const int* in_sizes, int n_in,
                              void* d_out, int out_size, void* d_ws, size_t ws_size,
                              hipStream_t stream) {
  (void)n_in; (void)out_size; (void)ws_size;
  const float* x    = (const float*)d_in[0];
  const float* c1w  = (const float*)d_in[1];
  const float* c1b  = (const float*)d_in[2];
  const float* a1   = (const float*)d_in[3];
  const float* c2w  = (const float*)d_in[4];
  const float* c2b  = (const float*)d_in[5];
  const float* a2   = (const float*)d_in[6];
  const float* c3w  = (const float*)d_in[7];
  const float* c3b  = (const float*)d_in[8];
  const float* a3   = (const float*)d_in[9];
  const float* c4w  = (const float*)d_in[10];
  const float* c4b  = (const float*)d_in[11];
  const float* a4   = (const float*)d_in[12];
  const float* d5w  = (const float*)d_in[13];
  const float* d5b  = (const float*)d_in[14];
  const float* a5   = (const float*)d_in[15];
  const float* d61w = (const float*)d_in[16];
  const float* d61b = (const float*)d_in[17];
  const float* d62w = (const float*)d_in[18];
  const float* d62b = (const float*)d_in[19];
  const float* d63w = (const float*)d_in[20];
  const float* d63b = (const float*)d_in[21];

  const int N = in_sizes[0] / (3 * 48 * 48);   // 2048

  char* ws = (char*)d_ws;
  size_t off = 0;
  auto take = [&](size_t bytes) {
    char* p = ws + off;
    off = (off + bytes + 255) & ~(size_t)255;
    return p;
  };
  _Float16* pool1 = (_Float16*)take((size_t)N * 32 * 529 * 2);
  _Float16* pool2 = (_Float16*)take((size_t)N * 64 * 100 * 2);
  _Float16* flat  = (_Float16*)take((size_t)N * 1152 * 2);
  _Float16* feat  = (_Float16*)take((size_t)N * 256 * 2);
  uint32_t* w1p = (uint32_t*)take((size_t)2  * 1  * 256 * 4);
  uint32_t* w2p = (uint32_t*)take((size_t)4  * 9  * 256 * 4);
  uint32_t* w3p = (uint32_t*)take((size_t)4  * 18 * 256 * 4);
  uint32_t* w4p = (uint32_t*)take((size_t)8  * 8  * 256 * 4);
  uint32_t* w5p = (uint32_t*)take((size_t)16 * 36 * 256 * 4);

  pack_b_kernel<<<2,   256, 0, stream>>>(c1w, w1p,  32,   27,  1);
  pack_b_kernel<<<36,  256, 0, stream>>>(c2w, w2p,  64,  288,  9);
  pack_b_kernel<<<72,  256, 0, stream>>>(c3w, w3p,  64,  576, 18);
  pack_b_kernel<<<64,  256, 0, stream>>>(c4w, w4p, 128,  256,  8);
  pack_b_kernel<<<576, 256, 0, stream>>>(d5w, w5p, 256, 1152, 36);

  conv1_pool_kernel<<<N * 3, 256, 0, stream>>>(x, w1p, c1b, a1, pool1);
  conv2_pool_kernel<<<N * 2, 256, 0, stream>>>(pool1, w2p, c2b, a2, pool2);
  conv34_kernel<<<N, 256, 0, stream>>>(pool2, w3p, c3b, a3, w4p, c4b, a4, flat);
  dense5_kernel<<<N / 8, 256, 0, stream>>>(flat, w5p, d5b, a5, feat);
  heads_kernel<<<N / 16, 256, 0, stream>>>(feat, d61w, d61b, d62w, d62b,
                                           d63w, d63b, (float*)d_out,
                                           N * 4, N * 14);
}
